// ImagineEntityAttnLayer_17231408792050
// MI455X (gfx1250) — matrix operand order repaired and checked by reading, not compiled, not run on hardware
//
#include <hip/hip_runtime.h>

// ---------------------------------------------------------------------------
// ImagineEntityAttnLayer fused kernel for MI455X (gfx1250, wave32, WMMA)
//
// B=32,T=32,A=8 -> 8192 groups; E=32 entities, D=256, H=8 heads, hd=32.
// - entities (256 MB fp32) read exactly once from HBM (~11us @ 23.3 TB/s),
//   staged via global_load_async_to_lds_b128 (ASYNCcnt path).
// - K/V projections (~69 GFLOP) on v_wmma_f32_16x16x32_bf16; bf16 weights
//   pre-converted into d_ws, L2-resident (0.5 MB << 192 MB L2). B fragments
//   for all 8 K-steps are held in registers and reused across both M-tiles
//   (halves L2 weight traffic to the per-block floor).
// - Attention (scores/softmax/attn/out-proj) is tiny -> VALU + wave32
//   shuffle reductions (one wave per head, lane = entity).
// - f32->bf16 uses the native __bf16 cast so the hardware packed convert is
//   used instead of a manual RNE bit sequence.
// ---------------------------------------------------------------------------

typedef __bf16 v16bf __attribute__((ext_vector_type(16)));
typedef float  v8f   __attribute__((ext_vector_type(8)));

constexpr int kD   = 256;   // model dim
constexpr int kE   = 32;    // entities per agent
constexpr int kH   = 8;     // heads
constexpr int kHD  = 32;    // head dim
constexpr int kGroups = 32 * 32 * 8;       // B*T*A
constexpr int kEntStride = kD + 8;         // bf16 LDS row stride (bank pad)
constexpr int kWsz = kD * kD;              // 65536 elems per weight matrix

__device__ __forceinline__ unsigned short f32_to_bf16(float f) {
  union { __bf16 b; unsigned short h; } v;
  v.b = (__bf16)f;                          // hardware RNE convert
  return v.h;
}
__device__ __forceinline__ float bf16_to_f32(unsigned short h) {
  union { unsigned int u; float f; } v; v.u = ((unsigned int)h) << 16;
  return v.f;
}

union Frag16 { v16bf v; uint4 q[2]; };

struct __align__(16) Smem {
  union {                                   // 33 KB region, used twice:
    float entf[kE][kD];                     //  phase 0: async fp32 staging
    struct {                                //  phase 1+: bf16 K/V tiles
      unsigned short K[kE][kEntStride];
      unsigned short V[kE][kEntStride];
    } kv;
  } u;
  unsigned short ent[kE][kEntStride];       // bf16 entity tile (A matrix)
  float qv[kD];
  float w[kH * kE];
  float attn[kD];
  int   mask[kE];
};

// --- prepass: fp32 weights -> bf16 in workspace (Wq | Wk | Wv | Wo) --------
__global__ void convert_weights_kernel(const float* __restrict__ Wq,
                                       const float* __restrict__ Wk,
                                       const float* __restrict__ Wv,
                                       const float* __restrict__ Wo,
                                       unsigned short* __restrict__ out) {
  int i = blockIdx.x * blockDim.x + threadIdx.x;   // 0 .. 65535
  out[i]            = f32_to_bf16(Wq[i]);
  out[i + kWsz]     = f32_to_bf16(Wk[i]);
  out[i + 2 * kWsz] = f32_to_bf16(Wv[i]);
  out[i + 3 * kWsz] = f32_to_bf16(Wo[i]);
}

// --- main fused kernel: one block (8 waves) per (b,t,a) group --------------
__global__ __launch_bounds__(256) void entity_attn_kernel(
    const float* __restrict__ entities,        // [G, E, D] fp32
    const int*   __restrict__ pre_mask,        // [G, E]
    const unsigned short* __restrict__ wbf,    // bf16 Wq|Wk|Wv|Wo
    float* __restrict__ out) {                 // [G, D]
  __shared__ Smem sm;
  const int tid  = threadIdx.x;
  const int lane = tid & 31;
  const int wave = tid >> 5;
  const int gid  = blockIdx.x;

  // ---- phase 0: async-copy entity tile (32x256 fp32) into LDS ------------
  {
    unsigned ldsbase =
        (unsigned)(size_t)(&sm.u.entf[0][0]);         // LDS offset (addr[31:0])
    unsigned long long gbase =
        (unsigned long long)(entities + (size_t)gid * (kE * kD));
    #pragma unroll
    for (int k = 0; k < 8; ++k) {
      int f = tid + k * 256;                          // 16-byte chunk index
      unsigned lds = ldsbase + (unsigned)f * 16u;
      unsigned long long ga = gbase + (unsigned long long)f * 16ull;
      asm volatile("global_load_async_to_lds_b128 %0, %1, off"
                   :: "v"(lds), "v"(ga) : "memory");
    }
    asm volatile("s_wait_asynccnt 0x0" ::: "memory");
  }
  if (tid < kE) sm.mask[tid] = pre_mask[gid * kE + tid];
  __syncthreads();

  // ---- phase 0b: fp32 -> bf16 into padded A-tile array -------------------
  #pragma unroll
  for (int k = 0; k < 8; ++k) {
    int f = tid + k * 256;
    int er = f >> 6;                                  // entity row
    int dc = (f & 63) * 4;                            // column
    float4 x = *reinterpret_cast<const float4*>(&sm.u.entf[er][dc]);
    union { unsigned short h[4]; unsigned long long u64; } t;
    t.h[0] = f32_to_bf16(x.x); t.h[1] = f32_to_bf16(x.y);
    t.h[2] = f32_to_bf16(x.z); t.h[3] = f32_to_bf16(x.w);
    *reinterpret_cast<unsigned long long*>(&sm.ent[er][dc]) = t.u64;
  }
  __syncthreads();   // entf fully consumed; region becomes K/V from here on

  // ---- phase 1: K/V projections via v_wmma_f32_16x16x32_bf16 -------------
  // 32 (proj, ntile) pairs; each wave owns 4. B fragments for all 8 K-steps
  // are kept in registers and reused across both M-tiles.
  const int hl = lane >> 4;                           // lane half (ISA K-split)
  const int ln = lane & 15;
  for (int pj = 0; pj < 4; ++pj) {
    int pair = wave * 4 + pj;
    int proj = pair >> 4;                             // 0 = K, 1 = V
    int nt   = pair & 15;
    const unsigned short* wrow =
        wbf + (proj ? 2 * kWsz : kWsz) + (nt * 16 + ln) * kD;   // W[n][*]
    Frag16 bfr[8];
    #pragma unroll
    for (int ks = 0; ks < 8; ++ks) {
      // B 32x16 bf16: lane = N, lane-half selects K 0-15 vs 16-31
      bfr[ks].q[0] = *reinterpret_cast<const uint4*>(&wrow[ks * 32 + hl * 16]);
      bfr[ks].q[1] = *reinterpret_cast<const uint4*>(&wrow[ks * 32 + hl * 16 + 8]);
    }
    unsigned short (*dst)[kEntStride] = proj ? sm.u.kv.V : sm.u.kv.K;
    #pragma unroll
    for (int mt = 0; mt < 2; ++mt) {
      const unsigned short* arow = &sm.ent[mt * 16 + ln][0];
      v8f acc = {};
      #pragma unroll
      for (int ks = 0; ks < 8; ++ks) {
        // A 16x32 bf16: lanes 0-15 hold K 0..15, 16-31 hold K 16..31
        Frag16 a;
        a.q[0] = *reinterpret_cast<const uint4*>(&arow[ks * 32 + hl * 16]);
        a.q[1] = *reinterpret_cast<const uint4*>(&arow[ks * 32 + hl * 16 + 8]);
        acc = __builtin_amdgcn_wmma_f32_16x16x32_bf16(
            false, a.v, false, bfr[ks].v, (short)0, acc, false, false);
      }
      #pragma unroll
      for (int r = 0; r < 8; ++r)      // C/D: VGPR r -> row hl*8+r, col = ln
        dst[mt * 16 + hl * 8 + r][nt * 16 + ln] = f32_to_bf16(acc[r]);
    }
  }

  // ---- q projection (entity 0 row): thread n computes q[n] ---------------
  {
    float acc = 0.f;
    const uint4* wq = reinterpret_cast<const uint4*>(wbf + tid * kD);
    const uint4* e0 = reinterpret_cast<const uint4*>(&sm.ent[0][0]);
    for (int c = 0; c < kD / 8; ++c) {
      uint4 wv4 = wq[c], ev4 = e0[c];
      const unsigned short* wp = (const unsigned short*)&wv4;
      const unsigned short* ep = (const unsigned short*)&ev4;
      #pragma unroll
      for (int j = 0; j < 8; ++j)
        acc += bf16_to_f32(ep[j]) * bf16_to_f32(wp[j]);
    }
    sm.qv[tid] = acc;
  }
  __syncthreads();

  // ---- scores + masked softmax: wave = head, lane = entity ---------------
  const int hh = wave;
  float s = 0.f;
  {
    const unsigned short* krow = &sm.u.kv.K[lane][hh * kHD];
    const float* qh = &sm.qv[hh * kHD];
    for (int j = 0; j < kHD; ++j) s += qh[j] * bf16_to_f32(krow[j]);
  }
  if (sm.mask[lane] != 0) s = -1e8f;   // reference: where(mask, FILL, score)
  float mx = s;
  for (int off = 16; off > 0; off >>= 1) mx = fmaxf(mx, __shfl_xor(mx, off, 32));
  float p = __expf(s - mx);
  float sum = p;
  for (int off = 16; off > 0; off >>= 1) sum += __shfl_xor(sum, off, 32);
  sm.w[tid] = p / sum;
  __syncthreads();

  // ---- attn[h][j] = sum_e w[h][e] * V[e][h*hd + j] ------------------------
  {
    float a = 0.f;
    const float* wh = &sm.w[hh * kE];
    for (int ee = 0; ee < kE; ++ee)
      a += wh[ee] * bf16_to_f32(sm.u.kv.V[ee][hh * kHD + lane]);
    sm.attn[tid] = a;
  }
  __syncthreads();

  // ---- output projection: out[n] = sum_d attn[d] * Wo[n][d] --------------
  {
    float o = 0.f;
    const uint4* wo = reinterpret_cast<const uint4*>(wbf + 3 * kWsz + tid * kD);
    for (int c = 0; c < kD / 8; ++c) {
      uint4 wv4 = wo[c];
      const unsigned short* wp = (const unsigned short*)&wv4;
      const float* ap = &sm.attn[c * 8];
      #pragma unroll
      for (int j = 0; j < 8; ++j)
        o += ap[j] * bf16_to_f32(wp[j]);
    }
    out[(size_t)gid * kD + tid] = o;
  }
}

extern "C" void kernel_launch(void* const* d_in, const int* in_sizes, int n_in,
                              void* d_out, int out_size, void* d_ws, size_t ws_size,
                              hipStream_t stream) {
  const float* entities = (const float*)d_in[0];
  const int*   pre_mask = (const int*)d_in[1];
  const float* Wq       = (const float*)d_in[2];
  const float* Wk       = (const float*)d_in[3];
  const float* Wv       = (const float*)d_in[4];
  const float* Wo       = (const float*)d_in[5];
  unsigned short* wbf = (unsigned short*)d_ws;   // 4 * 65536 bf16 = 512 KB
  float* out = (float*)d_out;

  convert_weights_kernel<<<kWsz / 256, 256, 0, stream>>>(Wq, Wk, Wv, Wo, wbf);
  entity_attn_kernel<<<kGroups, 256, 0, stream>>>(entities, pre_mask, wbf, out);
}
